// StandardAttention_75943611728135
// MI455X (gfx1250) — matrix operand order repaired and checked by reading, not compiled, not run on hardware
//
#include <hip/hip_runtime.h>

// ---------------------------------------------------------------------------
// Multi-head attention for gfx1250 (MI455X): bf16 WMMA everywhere, f32 accum,
// flash-style online softmax so S*S scores never touch HBM.
// ---------------------------------------------------------------------------

typedef __bf16 bf16_t;
typedef __attribute__((ext_vector_type(16))) __bf16 v16bf;
typedef __attribute__((ext_vector_type(8))) float v8f;

constexpr int kBatch = 2;
constexpr int kSeq   = 2048;
constexpr int kEmbed = 2048;
constexpr int kHeads = 16;
constexpr int kHDim  = 128;                       // 2048 / 16
constexpr float kScale = 0.0220970869120796f;     // EMBED^-0.5 (full embed dim!)

// ---------------------------------------------------------------------------
// WMMA helpers: fragment layouts per CDNA5 ISA 7.12.2 (wave32).
// ---------------------------------------------------------------------------

__device__ __forceinline__ v8f wmma_bf16(v16bf a, v16bf b, v8f c) {
  return __builtin_amdgcn_wmma_f32_16x16x32_bf16(
      /*neg_a=*/false, a, /*neg_b=*/false, b,
      /*c_mod=*/(short)0, c, /*reuse_a=*/false, /*reuse_b=*/false);
}

// A fragment: 16x32 bf16, rows [row0..row0+15], K [k0..k0+31] from an LDS
// tile stored row-major with row stride `ldk` (elements).
// Lane L<16  (M=L):    halfwords 0..7 -> K=0..7,   8..15 -> K=16..23
// Lane L>=16 (M=L-16): halfwords 0..7 -> K=8..15,  8..15 -> K=24..31
__device__ __forceinline__ v16bf load_frag_a(const bf16_t* base, int row0,
                                             int k0, int ldk) {
  const int lane = threadIdx.x & 31;
  const int lo = lane & 15, hi = lane >> 4;
  const bf16_t* p = base + (size_t)(row0 + lo) * ldk + k0 + 8 * hi;
  union { uint4 u[2]; v16bf v; } cvt;
  cvt.u[0] = *(const uint4*)(p);        // K = 8*hi + 0..7
  cvt.u[1] = *(const uint4*)(p + 16);   // K = 8*hi + 16..23
  return cvt.v;
}

// B fragment: 32x16 bf16 for output columns [n0..n0+15], K [k0..k0+31],
// from an LDS tile stored as [N][K] with row stride `ldk`.
// Lane L<16: N=L, K=0..7,16..23; Lane L>=16: N=L-16, K=8..15,24..31.
__device__ __forceinline__ v16bf load_frag_b(const bf16_t* base, int n0,
                                             int k0, int ldk) {
  const int lane = threadIdx.x & 31;
  const int lo = lane & 15, hi = lane >> 4;
  const bf16_t* p = base + (size_t)(n0 + lo) * ldk + k0 + 8 * hi;
  union { uint4 u[2]; v16bf v; } cvt;
  cvt.u[0] = *(const uint4*)(p);
  cvt.u[1] = *(const uint4*)(p + 16);
  return cvt.v;
}

// C/D layout: VGPR r, lanes 0..15 -> (M=r, N=lane); lanes 16..31 -> (M=8+r).
__device__ __forceinline__ void store_c_f32(float* base, int m0, int n0,
                                            int ldn, v8f acc) {
  const int lane = threadIdx.x & 31;
  const int lo = lane & 15, hi = lane >> 4;
#pragma unroll
  for (int r = 0; r < 8; ++r)
    base[(size_t)(m0 + r + 8 * hi) * ldn + n0 + lo] = acc[r];
}

// ---------------------------------------------------------------------------
// fp32 -> bf16 conversion (vector of 4 per thread)
// ---------------------------------------------------------------------------
__global__ void cvt_f32_to_bf16(const float* __restrict__ in,
                                bf16_t* __restrict__ out, int n4) {
  int i = blockIdx.x * blockDim.x + threadIdx.x;
  if (i >= n4) return;
  float4 v = ((const float4*)in)[i];
  bf16_t o4[4] = {(bf16_t)v.x, (bf16_t)v.y, (bf16_t)v.z, (bf16_t)v.w};
  *(uint2*)&out[4 * i] = *(const uint2*)o4;
}

// ---------------------------------------------------------------------------
// GEMM: Y[M,N] = A[M,K](bf16) * W[N,K]^T(bf16) + bias[N]
// torch Linear weight layout [N,K] maps directly onto the B-fragment [N][K].
// Block tile 128x128, BK=32; 8 waves in 4(M) x 2(N); wave tile 32x64.
// Software-pipelined: next K-tile staged global->VGPR during the WMMA burst,
// committed to LDS at the next iteration's top barrier. GL2 prefetch one
// further K-step ahead.
// ---------------------------------------------------------------------------
template <bool STORE_BF16>
__global__ __launch_bounds__(256) void gemm_bias(
    const bf16_t* __restrict__ A, const bf16_t* __restrict__ W,
    const float* __restrict__ bias, void* __restrict__ Yv, int M, int N,
    int Kd) {
  __shared__ bf16_t sA[128 * 40];   // 128 x 32, row stride 40 (80B, 16B mult)
  __shared__ bf16_t sB[128 * 40];   // 128 x 32

  const int tid = threadIdx.x;
  const int lane = tid & 31;
  const int w = tid >> 5;
  const int lo = lane & 15, hi = lane >> 4;
  const int m0 = blockIdx.y * 128;
  const int n0 = blockIdx.x * 128;
  const int tm = (w >> 1) * 32;   // wave M offset in tile (4 waves)
  const int tn = (w & 1) * 64;    // wave N offset in tile (2 waves)

  // Per-thread staging slots: 2 uint4 for A tile, 2 for B tile.
  // idx = tid + j*256 over 512 uint4s; row = idx>>2, kchunk = (idx&3)*8.
  const int r0 = (tid + 0) >> 2, c0 = ((tid + 0) & 3) * 8;
  const int r1 = (tid + 256) >> 2, c1 = ((tid + 256) & 3) * 8;

  v8f acc[2][4] = {};
  uint4 ra0, ra1, rb0, rb1;

  // Prologue: stage K-tile 0.
  ra0 = *(const uint4*)(A + (size_t)(m0 + r0) * Kd + c0);
  ra1 = *(const uint4*)(A + (size_t)(m0 + r1) * Kd + c1);
  rb0 = *(const uint4*)(W + (size_t)(n0 + r0) * Kd + c0);
  rb1 = *(const uint4*)(W + (size_t)(n0 + r1) * Kd + c1);

  for (int k0 = 0; k0 < Kd; k0 += 32) {
    __syncthreads();  // previous iter's fragment reads complete
    *(uint4*)&sA[r0 * 40 + c0] = ra0;
    *(uint4*)&sA[r1 * 40 + c1] = ra1;
    *(uint4*)&sB[r0 * 40 + c0] = rb0;
    *(uint4*)&sB[r1 * 40 + c1] = rb1;
    __syncthreads();  // tile visible

    // Stage next K-tile while the WMMAs run; prefetch one more ahead to GL2.
    if (k0 + 32 < Kd) {
      const int kn = k0 + 32;
      ra0 = *(const uint4*)(A + (size_t)(m0 + r0) * Kd + kn + c0);
      ra1 = *(const uint4*)(A + (size_t)(m0 + r1) * Kd + kn + c1);
      rb0 = *(const uint4*)(W + (size_t)(n0 + r0) * Kd + kn + c0);
      rb1 = *(const uint4*)(W + (size_t)(n0 + r1) * Kd + kn + c1);
      if (k0 + 64 < Kd) {
        __builtin_prefetch(A + (size_t)(m0 + r0) * Kd + k0 + 64 + c0, 0, 3);
        __builtin_prefetch(W + (size_t)(n0 + r0) * Kd + k0 + 64 + c0, 0, 3);
      }
    }

    v16bf a0 = load_frag_a(sA, tm, 0, 40);
    v16bf a1 = load_frag_a(sA, tm + 16, 0, 40);
#pragma unroll
    for (int j = 0; j < 4; ++j) {
      v16bf bj = load_frag_b(sB, tn + j * 16, 0, 40);
      acc[0][j] = wmma_bf16(a0, bj, acc[0][j]);
      acc[1][j] = wmma_bf16(a1, bj, acc[1][j]);
    }
  }

#pragma unroll
  for (int i = 0; i < 2; ++i)
#pragma unroll
    for (int j = 0; j < 4; ++j)
#pragma unroll
      for (int r = 0; r < 8; ++r) {
        int row = m0 + tm + i * 16 + r + 8 * hi;
        int col = n0 + tn + j * 16 + lo;
        float val = acc[i][j][r] + bias[col];
        if (STORE_BF16)
          ((bf16_t*)Yv)[(size_t)row * N + col] = (bf16_t)val;
        else
          ((float*)Yv)[(size_t)row * N + col] = val;
      }
}

// ---------------------------------------------------------------------------
// Flash attention: one workgroup per (b, h, 64-row query tile).
// Key/value tiles of 64; online softmax with running max/sum; O in registers
// distributed across 8 waves (2Mx4N grid, 32x32 per wave).
// ---------------------------------------------------------------------------
__global__ __launch_bounds__(256) void attn_kernel(
    const bf16_t* __restrict__ Q, const bf16_t* __restrict__ K,
    const bf16_t* __restrict__ V, bf16_t* __restrict__ O) {
  __shared__ bf16_t sQ[64 * 136];    // 64 x 128 bf16 (stride 136)
  __shared__ bf16_t sK[64 * 136];    // 64 x 128 bf16
  __shared__ bf16_t sVt[128 * 72];   // V transposed: [d][key], stride 72
  __shared__ float sS[64 * 68];      // raw scores f32
  __shared__ bf16_t sP[64 * 72];     // probabilities bf16
  __shared__ float sRed[64 * 4];
  __shared__ float sM[64], sL[64], sAlpha[64];

  const int tid = threadIdx.x;
  const int lane = tid & 31;
  const int w = tid >> 5;
  const int lo = lane & 15, hi = lane >> 4;

  const int bh = blockIdx.y;
  const int b = bh >> 4, h = bh & 15;
  const int q0 = blockIdx.x * 64;

  // Load the Q tile (64 rows x 128 d) once.
#pragma unroll
  for (int j = 0; j < 4; ++j) {
    int idx = tid + j * 256;
    int row = idx >> 4;
    int c = (idx & 15) * 8;
    *(uint4*)&sQ[row * 136 + c] =
        *(const uint4*)(Q + (size_t)(b * kSeq + q0 + row) * kEmbed +
                        h * kHDim + c);
  }
  if (tid < 64) { sM[tid] = -1e30f; sL[tid] = 0.0f; }

  v8f o[2][2] = {};
  const int ow_m = (w >> 2) * 32;   // O tile: 2(M) x 4(N) wave grid
  const int ow_n = (w & 3) * 32;
  const int sm = (w >> 1) * 16;     // scores tile: 4(M) x 2(N) wave grid
  const int sn = (w & 1) * 32;
  const int srow = tid >> 2;        // softmax: 4 threads per row
  const int sc4 = tid & 3;

  for (int t = 0; t < kSeq / 64; ++t) {
    const int kb = t * 64;
    __syncthreads();  // previous iter's reads of sK/sVt/sP done

    // Load K tile [key][d] and V tile transposed [d][key].
#pragma unroll
    for (int j = 0; j < 4; ++j) {
      int idx = tid + j * 256;
      int row = idx >> 4;
      int c = (idx & 15) * 8;
      const size_t gro = (size_t)(b * kSeq + kb + row) * kEmbed + h * kHDim + c;
      *(uint4*)&sK[row * 136 + c] = *(const uint4*)(K + gro);
      uint4 vv = *(const uint4*)(V + gro);
      const bf16_t* pv = (const bf16_t*)&vv;
#pragma unroll
      for (int e = 0; e < 8; ++e) sVt[(c + e) * 72 + row] = pv[e];
    }
    __syncthreads();

    // Scores S = Q * K^T  (K-dim 128 = 4 WMMA k-steps); wave does 16x32.
    v8f s0 = {}, s1 = {};
#pragma unroll
    for (int kk = 0; kk < 4; ++kk) {
      v16bf a = load_frag_a(sQ, sm, kk * 32, 136);
      v16bf bb0 = load_frag_b(sK, sn, kk * 32, 136);
      v16bf bb1 = load_frag_b(sK, sn + 16, kk * 32, 136);
      s0 = wmma_bf16(a, bb0, s0);
      s1 = wmma_bf16(a, bb1, s1);
    }
    store_c_f32(sS, sm, sn, 68, s0);
    store_c_f32(sS, sm, sn + 16, 68, s1);
    __syncthreads();

    // Row max (scaled domain), partial across 4 chunks of 16 cols.
    {
      const float* rowp = &sS[srow * 68 + sc4 * 16];
      float mx = -1e30f;
#pragma unroll
      for (int i = 0; i < 16; ++i) mx = fmaxf(mx, rowp[i]);
      sRed[srow * 4 + sc4] = mx;
    }
    __syncthreads();
    if (sc4 == 0) {
      float mt = fmaxf(fmaxf(sRed[srow * 4 + 0], sRed[srow * 4 + 1]),
                       fmaxf(sRed[srow * 4 + 2], sRed[srow * 4 + 3])) * kScale;
      float mo = sM[srow];
      float mn = fmaxf(mo, mt);
      sAlpha[srow] = __expf(mo - mn);
      sM[srow] = mn;
    }
    __syncthreads();

    // P = exp(S*scale - m), bf16 into LDS; partial row sums.
    {
      float mn = sM[srow];
      const float* rowp = &sS[srow * 68 + sc4 * 16];
      bf16_t* pp = &sP[srow * 72 + sc4 * 16];
      float sum = 0.0f;
#pragma unroll
      for (int i = 0; i < 16; ++i) {
        float p = __expf(rowp[i] * kScale - mn);
        pp[i] = (bf16_t)p;
        sum += p;
      }
      sRed[srow * 4 + sc4] = sum;
    }
    __syncthreads();
    if (sc4 == 0) {
      float sum = sRed[srow * 4 + 0] + sRed[srow * 4 + 1] +
                  sRed[srow * 4 + 2] + sRed[srow * 4 + 3];
      sL[srow] = sL[srow] * sAlpha[srow] + sum;
    }

    // Rescale running O by alpha (per row).
#pragma unroll
    for (int i = 0; i < 2; ++i)
#pragma unroll
      for (int r = 0; r < 8; ++r) {
        float alpha = sAlpha[ow_m + i * 16 + r + 8 * hi];
        o[i][0][r] *= alpha;
        o[i][1][r] *= alpha;
      }

    // O += P * V   (K-dim 64 = 2 WMMA k-steps); wave owns 32x32 of O.
#pragma unroll
    for (int kk = 0; kk < 2; ++kk) {
      v16bf a0 = load_frag_a(sP, ow_m, kk * 32, 72);
      v16bf a1 = load_frag_a(sP, ow_m + 16, kk * 32, 72);
      v16bf bb0 = load_frag_b(sVt, ow_n, kk * 32, 72);
      v16bf bb1 = load_frag_b(sVt, ow_n + 16, kk * 32, 72);
      o[0][0] = wmma_bf16(a0, bb0, o[0][0]);
      o[0][1] = wmma_bf16(a0, bb1, o[0][1]);
      o[1][0] = wmma_bf16(a1, bb0, o[1][0]);
      o[1][1] = wmma_bf16(a1, bb1, o[1][1]);
    }
  }
  __syncthreads();  // sL final

  // Normalize by l and emit attn (bf16) at [b, s, h*128 + d].
#pragma unroll
  for (int i = 0; i < 2; ++i)
#pragma unroll
    for (int j = 0; j < 2; ++j)
#pragma unroll
      for (int r = 0; r < 8; ++r) {
        int row = ow_m + i * 16 + r + 8 * hi;
        int col = ow_n + j * 16 + lo;
        float val = o[i][j][r] / sL[row];
        O[(size_t)(b * kSeq + q0 + row) * kEmbed + h * kHDim + col] =
            (bf16_t)val;
      }
}

// ---------------------------------------------------------------------------
// Host-side orchestration.
// ---------------------------------------------------------------------------
extern "C" void kernel_launch(void* const* d_in, const int* in_sizes, int n_in,
                              void* d_out, int out_size, void* d_ws,
                              size_t ws_size, hipStream_t stream) {
  (void)in_sizes; (void)n_in; (void)out_size; (void)ws_size;
  const float* x  = (const float*)d_in[0];
  const float* wq = (const float*)d_in[1];
  const float* bq = (const float*)d_in[2];
  const float* wk = (const float*)d_in[3];
  const float* bk = (const float*)d_in[4];
  const float* wv = (const float*)d_in[5];
  const float* bv = (const float*)d_in[6];
  const float* wo = (const float*)d_in[7];
  const float* bo = (const float*)d_in[8];

  const size_t M = (size_t)kBatch * kSeq;  // 4096
  const size_t E = kEmbed;                 // 2048

  char* ws = (char*)d_ws;
  size_t off = 0;
  auto carve = [&](size_t bytes) -> char* {
    char* p = ws + off;
    off += (bytes + 255) & ~(size_t)255;
    return p;
  };
  bf16_t* xb  = (bf16_t*)carve(M * E * 2);
  bf16_t* wqb = (bf16_t*)carve(E * E * 2);
  bf16_t* wkb = (bf16_t*)carve(E * E * 2);
  bf16_t* wvb = (bf16_t*)carve(E * E * 2);
  bf16_t* wob = (bf16_t*)carve(E * E * 2);
  bf16_t* Qb  = (bf16_t*)carve(M * E * 2);
  bf16_t* Kb  = (bf16_t*)carve(M * E * 2);
  bf16_t* Vb  = (bf16_t*)carve(M * E * 2);
  bf16_t* Ab  = (bf16_t*)carve(M * E * 2);

  auto cvt = [&](const float* src, bf16_t* dst, size_t n) {
    int n4 = (int)(n / 4);
    cvt_f32_to_bf16<<<(n4 + 255) / 256, 256, 0, stream>>>(src, dst, n4);
  };
  cvt(x, xb, M * E);
  cvt(wq, wqb, E * E);
  cvt(wk, wkb, E * E);
  cvt(wv, wvb, E * E);
  cvt(wo, wob, E * E);

  dim3 gg((unsigned)(E / 128), (unsigned)(M / 128));  // 16 x 32
  gemm_bias<true><<<gg, 256, 0, stream>>>(xb, wqb, bq, Qb, (int)M, (int)E, (int)E);
  gemm_bias<true><<<gg, 256, 0, stream>>>(xb, wkb, bk, Kb, (int)M, (int)E, (int)E);
  gemm_bias<true><<<gg, 256, 0, stream>>>(xb, wvb, bv, Vb, (int)M, (int)E, (int)E);

  attn_kernel<<<dim3(kSeq / 64, kBatch * kHeads), 256, 0, stream>>>(Qb, Kb, Vb, Ab);

  gemm_bias<false><<<gg, 256, 0, stream>>>(Ab, wob, bo, d_out, (int)M, (int)E, (int)E);
}
